// SpikingTemporalEncoder_11484742550220
// MI455X (gfx1250) — matrix operand order repaired and checked by reading, compiled, not run on hardware
//
#include <hip/hip_runtime.h>
#include <hip/hip_bf16.h>
#include <math.h>

// ---------------------------------------------------------------------------
// SpikingTemporalEncoder forward for MI455X (gfx1250, wave32, WMMA).
// Memory-bound problem (~8 GB traffic vs ~550 GFLOP) -> bf16 WMMA GEMMs with
// f32 accumulate, fused bias, separate streaming LN / LIF kernels.
// ---------------------------------------------------------------------------

typedef __attribute__((ext_vector_type(16))) __bf16 v16bf;
typedef __attribute__((ext_vector_type(8)))  __bf16 v8bf;
typedef __attribute__((ext_vector_type(8)))  float  v8f;

#define B_   256
#define T_   256
#define IN_  4
#define HP_  64
#define E_   512
#define H_   1024
#define D_   4
#define M_   (B_*T_)   // 65536 rows

__device__ __forceinline__ v16bf cat8(v8bf lo, v8bf hi) {
  return __builtin_shufflevector(lo, hi, 0,1,2,3,4,5,6,7,8,9,10,11,12,13,14,15);
}

// ---------------------------------------------------------------------------
// Repack W[K,N] (f32, row-major) into bf16 B-fragments for wmma_f32_16x16x32_bf16.
// Packed layout: [kt][nt][lane][16 contiguous bf16], where per ISA B layout:
//   lanes 0-15  : N = nt*16+lane,    K = kt*32 + {0..7, 16..23}
//   lanes 16-31 : N = nt*16+lane-16, K = kt*32 + {8..15, 24..31}
// ---------------------------------------------------------------------------
__global__ void pack_b_kernel(const float* __restrict__ W, __bf16* __restrict__ out,
                              int K, int N, int nMat) {
  long gid = (long)blockIdx.x * blockDim.x + threadIdx.x;
  long per = (long)K * N;
  if (gid >= per * nMat) return;
  int  m    = (int)(gid / per);
  long rem  = gid - (long)m * per;
  int  j    = (int)(rem & 15);
  int  lane = (int)((rem >> 4) & 31);
  long tile = rem >> 9;
  int  nTiles = N >> 4;
  int  nt = (int)(tile % nTiles);
  int  kt = (int)(tile / nTiles);
  int  k  = kt * 32 + ((lane >> 4) << 3) + (j & 7) + ((j & 8) << 1);   // lane<16 -> +0 ; lane>=16 -> +8
  int  n  = nt * 16 + (lane & 15);
  out[(long)m * per + rem] = (__bf16)W[(long)m * per + (long)k * N + n];
}

// ---------------------------------------------------------------------------
// Input projector: x = gelu(hist @ pW1 + pb1) @ pW2 + pb2 ; writes f32 + bf16.
// 4 rows per block: threads 0..255 = 4 rows x 64 hidden, then 2048 outputs.
// ---------------------------------------------------------------------------
__global__ __launch_bounds__(256)
void projector_kernel(const float* __restrict__ hist, const float* __restrict__ pW1,
                      const float* __restrict__ pb1,  const float* __restrict__ pW2,
                      const float* __restrict__ pb2,
                      float* __restrict__ xf, __bf16* __restrict__ xb) {
  __shared__ float hbuf[4][HP_];
  const int rowBase = blockIdx.x * 4;
  const int tid = threadIdx.x;
  {
    int r = tid >> 6;
    int h = tid & 63;
    const float* xin = hist + (size_t)(rowBase + r) * IN_;
    float acc = pb1[h];
    #pragma unroll
    for (int i = 0; i < IN_; ++i) acc += xin[i] * pW1[i * HP_ + h];
    hbuf[r][h] = 0.5f * acc * (1.0f + erff(acc * 0.70710678118654752f)); // exact GELU
  }
  __syncthreads();
  for (int o = tid; o < 4 * E_; o += 256) {
    int r = o >> 9;
    int e = o & (E_ - 1);
    const float* hb = hbuf[r];
    float acc = pb2[e];
    #pragma unroll 8
    for (int h = 0; h < HP_; ++h) acc += hb[h] * pW2[h * E_ + e];
    size_t idx = (size_t)(rowBase + r) * E_ + e;
    xf[idx] = acc;
    xb[idx] = (__bf16)acc;
  }
}

// ---------------------------------------------------------------------------
// bf16 WMMA GEMM: C[M,N] = A[M,K] @ Bpacked + bias.  256 thr = 8 waves.
// Block tile 128x128, wave tile 32x64 (2 M-frags x 4 N-frags), K step 32.
// ---------------------------------------------------------------------------
__global__ __launch_bounds__(256)
void gemm_bf16_wmma(const __bf16* __restrict__ A, const __bf16* __restrict__ Bp,
                    const float* __restrict__ bias, float* __restrict__ C,
                    int N, int K) {
  const int tid  = threadIdx.x;
  const int lane = tid & 31;
  const int wave = tid >> 5;
  const int wm   = wave & 3;          // 4 waves along M
  const int wn   = wave >> 2;         // 2 waves along N
  const int mBase = blockIdx.y * 128 + wm * 32;
  const int nBase = blockIdx.x * 128 + wn * 64;
  const int l15 = lane & 15;
  const int hi8 = (lane >> 4) * 8;    // 0 or 8 (A/C half-lane K/M offset)
  const int nTiles = N >> 4;

  const v8f vzero = {0.f,0.f,0.f,0.f,0.f,0.f,0.f,0.f};
  v8f acc[2][4];
  #pragma unroll
  for (int i = 0; i < 2; ++i)
    #pragma unroll
    for (int j = 0; j < 4; ++j) acc[i][j] = vzero;

  // A 16-bit layout: lanes 0-15 hold row M0+lane, K {0..7, 16..23};
  //                  lanes 16-31 hold row M0+lane-16, K {8..15, 24..31}.
  const __bf16* arow0 = A + (size_t)(mBase + l15) * K + hi8;
  const __bf16* arow1 = A + (size_t)(mBase + 16 + l15) * K + hi8;
  const __bf16* bbase = Bp + ((size_t)(nBase >> 4) * 32 + lane) * 16;
  const size_t  bkstride = (size_t)nTiles * 512;   // elements per kt block

  for (int k0 = 0; k0 < K; k0 += 32) {
    __builtin_prefetch(arow0 + k0 + 64, 0, 0);   // global_prefetch next tiles
    v8bf a0lo = *(const v8bf*)(arow0 + k0);
    v8bf a0hi = *(const v8bf*)(arow0 + k0 + 16);
    v8bf a1lo = *(const v8bf*)(arow1 + k0);
    v8bf a1hi = *(const v8bf*)(arow1 + k0 + 16);
    v16bf a0 = cat8(a0lo, a0hi);
    v16bf a1 = cat8(a1lo, a1hi);
    const __bf16* bk = bbase + (size_t)(k0 >> 5) * bkstride;
    #pragma unroll
    for (int j = 0; j < 4; ++j) {
      v16bf b = *(const v16bf*)(bk + (size_t)j * 512);
      acc[0][j] = __builtin_amdgcn_wmma_f32_16x16x32_bf16(
          false, a0, false, b, (short)0, acc[0][j], false, false);
      acc[1][j] = __builtin_amdgcn_wmma_f32_16x16x32_bf16(
          false, a1, false, b, (short)0, acc[1][j], false, false);
    }
  }

  // C/D layout: VGPR r -> row = mBase + i*16 + hi8 + r ; col = nBase + j*16 + l15
  #pragma unroll
  for (int i = 0; i < 2; ++i) {
    int mB = mBase + i * 16 + hi8;
    #pragma unroll
    for (int j = 0; j < 4; ++j) {
      int col = nBase + j * 16 + l15;
      float bv = bias[col];
      float* crow = C + (size_t)mB * N + col;
      #pragma unroll
      for (int r = 0; r < 8; ++r)
        crow[(size_t)r * N] = acc[i][j][r] + bv;
    }
  }
}

// ---------------------------------------------------------------------------
// Row LayerNorm (in place): block per row, 256-thread tree reduction in LDS.
// ---------------------------------------------------------------------------
__global__ __launch_bounds__(256)
void layernorm_kernel(float* __restrict__ Y, const float* __restrict__ g,
                      const float* __restrict__ b, int N) {
  __shared__ float s1[256], s2[256];
  float* y = Y + (size_t)blockIdx.x * N;
  const int tid = threadIdx.x;
  float sum = 0.f, sq = 0.f;
  for (int i = tid; i < N; i += 256) { float v = y[i]; sum += v; sq += v * v; }
  s1[tid] = sum; s2[tid] = sq;
  __syncthreads();
  #pragma unroll
  for (int s = 128; s > 0; s >>= 1) {
    if (tid < s) { s1[tid] += s1[tid + s]; s2[tid] += s2[tid + s]; }
    __syncthreads();
  }
  float mean = s1[0] / N;
  float var  = s2[0] / N - mean * mean;
  float inv  = rsqrtf(var + 1e-5f);
  for (int i = tid; i < N; i += 256)
    y[i] = (y[i] - mean) * inv * g[i] + b[i];
}

// ---------------------------------------------------------------------------
// LIF scan (tau=2, vth=1, hard reset): thread per (b, feature) chain over T.
// Writes spikes as bf16 (exact {0,1}) for the following GEMM.
// ---------------------------------------------------------------------------
__global__ __launch_bounds__(256)
void lif_spike_kernel(const float* __restrict__ Y, __bf16* __restrict__ S, int Nfeat) {
  int gid = blockIdx.x * blockDim.x + threadIdx.x;
  if (gid >= B_ * Nfeat) return;
  int b = gid / Nfeat;
  int f = gid - b * Nfeat;
  size_t base = (size_t)b * T_ * Nfeat + f;
  float v = 0.f;
  for (int t = 0; t < T_; ++t) {
    size_t idx = base + (size_t)t * Nfeat;
    v = 0.5f * (v + Y[idx]);              // v += (x - v)/tau, tau = 2
    bool fire = (v >= 1.0f);
    S[idx] = (__bf16)(fire ? 1.0f : 0.0f);
    v = fire ? 0.f : v;                   // hard reset
  }
}

// Second LIF fused with residual: x_new = res + spike; also emit bf16 copy.
__global__ __launch_bounds__(256)
void lif_residual_kernel(const float* __restrict__ Z, float* __restrict__ xf,
                         __bf16* __restrict__ xb) {
  int gid = blockIdx.x * blockDim.x + threadIdx.x;
  if (gid >= B_ * E_) return;
  int b = gid >> 9;                       // / E_
  int e = gid & (E_ - 1);
  size_t base = (size_t)b * T_ * E_ + e;
  float v = 0.f;
  for (int t = 0; t < T_; ++t) {
    size_t idx = base + (size_t)t * E_;
    v = 0.5f * (v + Z[idx]);
    bool fire = (v >= 1.0f);
    float s = fire ? 1.0f : 0.0f;
    v = fire ? 0.f : v;
    float xn = xf[idx] + s;
    xf[idx] = xn;
    xb[idx] = (__bf16)xn;
  }
}

__global__ void extract_last_kernel(const float* __restrict__ xf, float* __restrict__ out) {
  int gid = blockIdx.x * blockDim.x + threadIdx.x;
  if (gid >= B_ * E_) return;
  int b = gid >> 9;
  int e = gid & (E_ - 1);
  out[gid] = xf[(size_t)(b * T_ + (T_ - 1)) * E_ + e];
}

// ---------------------------------------------------------------------------
extern "C" void kernel_launch(void* const* d_in, const int* in_sizes, int n_in,
                              void* d_out, int out_size, void* d_ws, size_t ws_size,
                              hipStream_t stream) {
  (void)in_sizes; (void)n_in; (void)out_size; (void)ws_size;
  const float* hist  = (const float*)d_in[0];
  const float* pW1   = (const float*)d_in[1];
  const float* pb1   = (const float*)d_in[2];
  const float* pW2   = (const float*)d_in[3];
  const float* pb2   = (const float*)d_in[4];
  const float* fc1_w = (const float*)d_in[5];
  const float* fc1_b = (const float*)d_in[6];
  const float* ln1_g = (const float*)d_in[7];
  const float* ln1_b = (const float*)d_in[8];
  const float* fc2_w = (const float*)d_in[9];
  const float* fc2_b = (const float*)d_in[10];
  const float* ln2_g = (const float*)d_in[11];
  const float* ln2_b = (const float*)d_in[12];

  // Workspace carve-out (~585 MB total, 256B aligned slices).
  char* ws = (char*)d_ws;
  size_t off = 0;
  auto take = [&](size_t bytes) -> char* {
    char* p = ws + off;
    off = (off + bytes + 255) & ~(size_t)255;
    return p;
  };
  float*  xf  = (float*) take((size_t)M_ * E_ * 4);      // residual stream f32
  __bf16* xb  = (__bf16*)take((size_t)M_ * E_ * 2);      // GEMM1 A input bf16
  float*  Y   = (float*) take((size_t)M_ * H_ * 4);      // GEMM1 out (Z aliases it)
  __bf16* S   = (__bf16*)take((size_t)M_ * H_ * 2);      // spikes bf16 (GEMM2 A)
  __bf16* W1p = (__bf16*)take((size_t)D_ * E_ * H_ * 2); // packed fc1 weights
  __bf16* W2p = (__bf16*)take((size_t)D_ * H_ * E_ * 2); // packed fc2 weights
  float*  Z   = Y;  // GEMM2 output reuses Y's storage (Y dead once S is built)

  // One-time weight repack (per launch; deterministic).
  {
    long tot = (long)D_ * E_ * H_;
    int blocks = (int)((tot + 255) / 256);
    pack_b_kernel<<<blocks, 256, 0, stream>>>(fc1_w, W1p, E_, H_, D_);
    pack_b_kernel<<<blocks, 256, 0, stream>>>(fc2_w, W2p, H_, E_, D_);
  }

  projector_kernel<<<M_ / 4, 256, 0, stream>>>(hist, pW1, pb1, pW2, pb2, xf, xb);

  for (int d = 0; d < D_; ++d) {
    gemm_bf16_wmma<<<dim3(H_ / 128, M_ / 128), 256, 0, stream>>>(
        xb, W1p + (size_t)d * E_ * H_, fc1_b + d * H_, Y, H_, E_);
    layernorm_kernel<<<M_, 256, 0, stream>>>(Y, ln1_g + d * H_, ln1_b + d * H_, H_);
    lif_spike_kernel<<<(B_ * H_) / 256, 256, 0, stream>>>(Y, S, H_);

    gemm_bf16_wmma<<<dim3(E_ / 128, M_ / 128), 256, 0, stream>>>(
        S, W2p + (size_t)d * H_ * E_, fc2_b + d * E_, Z, E_, H_);
    layernorm_kernel<<<M_, 256, 0, stream>>>(Z, ln2_g + d * E_, ln2_b + d * E_, E_);
    lif_residual_kernel<<<(B_ * E_) / 256, 256, 0, stream>>>(Z, xf, xb);
  }

  extract_last_kernel<<<(B_ * E_) / 256, 256, 0, stream>>>(xf, (float*)d_out);
}
